// GPT_48627619725753
// MI455X (gfx1250) — matrix operand order repaired and checked by reading, not compiled, not run on hardware
//
#include <hip/hip_runtime.h>
#include <hip/hip_bf16.h>
#include <stdint.h>

typedef __bf16 bf16;
typedef __attribute__((ext_vector_type(16))) __bf16 v16bf;
typedef __attribute__((ext_vector_type(8)))  float   v8f;
typedef unsigned int u32x4 __attribute__((ext_vector_type(4)));

union FragAB { u32x4 q[2]; v16bf v; };

// CDNA5 async global->LDS copy (ASYNCcnt-tracked), bypasses VGPR staging.
__device__ __forceinline__ void async_load_b128(unsigned lds_off, const void* gptr) {
    asm volatile("global_load_async_to_lds_b128 %0, %1, off"
                 :: "v"(lds_off), "v"((unsigned long long)(uintptr_t)gptr)
                 : "memory");
}
__device__ __forceinline__ void wait_async0() {
    asm volatile("s_wait_asynccnt 0" ::: "memory");
}

// ---------------------------------------------------------------------------
// Generic bf16 GEMM, B stored transposed [N,K] ("NT"), fp32 accumulate (WMMA).
// C[m,n] = act( scale * sum_k A[m,k]*B[n,k] + bias[n] ) + resid[m,n]
// Batched over blockIdx.z: z -> (zb = z/H, zh = z%H) with per-b / per-h strides.
// Tile: 128x128x64 per 256-thread block; wave grid 2x4; wave tile 64x32.
// Double-buffered LDS, async global->LDS loads overlap the WMMA stream.
// Ragged N handled by clamping B rows (dead columns never stored).
// causal: 0 = none, 1 = skip blocks fully above diagonal (QK^T scores),
//         2 = limit K loop to the causal row-block bound (P@V).
// ---------------------------------------------------------------------------
__global__ __launch_bounds__(256) void gpt_gemm_bf16_nt(
    const bf16* __restrict__ A, int lda, long sAb, long sAh,
    const bf16* __restrict__ B, int ldb, long sBb, long sBh,
    float* __restrict__ outF, bf16* __restrict__ outB,
    int ldc, long sCb, long sCh,
    const float* __restrict__ resid, const float* __restrict__ bias,
    int M, int N, int K, float scale, int act, int H, int causal)
{
    const int m0 = blockIdx.y * 128;
    const int n0 = blockIdx.x * 128;

    // Causal score blocks entirely above the diagonal are never read: skip.
    if (causal == 1 && n0 >= m0 + 128) return;
    // Causal P@V: K-tiles beyond the row-block bound multiply zeros: clamp.
    int Kend = K;
    if (causal == 2 && m0 + 128 < K) Kend = m0 + 128;

    __shared__ __align__(16) bf16 Alds[2][128 * 72];
    __shared__ __align__(16) bf16 Blds[2][128 * 72];

    const int tid = threadIdx.x;
    const int z  = blockIdx.z;
    const int zb = z / H, zh = z % H;
    A += zb * sAb + zh * sAh;
    B += zb * sBb + zh * sBh;
    const long c0 = zb * sCb + zh * sCh;

    // --- staging indices: each thread owns row tid>>1, 32 K elems (4 x b128)
    const int lrow = tid >> 1;
    const int lk   = (tid & 1) * 32;
    const bf16* Aptr = A + (long)(m0 + lrow) * lda + lk;
    const int brow = (n0 + lrow < N) ? (n0 + lrow) : (N - 1);   // clamp ragged N
    const bf16* Bptr = B + (long)brow * ldb + lk;

    const unsigned aoff = (unsigned)(uintptr_t)(&Alds[0][lrow * 72 + lk]);
    const unsigned boff = (unsigned)(uintptr_t)(&Blds[0][lrow * 72 + lk]);
    const unsigned bufstride = 128u * 72u * 2u;   // bytes per LDS buffer

    auto issue = [&](int kt, int b) {
        const unsigned ab = aoff + (unsigned)b * bufstride;
        const unsigned bb = boff + (unsigned)b * bufstride;
        const bf16* ap = Aptr + kt;
        const bf16* bp = Bptr + kt;
#pragma unroll
        for (int i = 0; i < 4; i++) {
            async_load_b128(ab + 16u * i, ap + 8 * i);
            async_load_b128(bb + 16u * i, bp + 8 * i);
        }
    };

    // --- wave / lane decomposition
    const int wave = tid >> 5;
    const int lane = tid & 31;
    const int wm = (wave >> 2) * 64;   // 0 or 64
    const int wn = (wave & 3) * 32;    // 0,32,64,96
    const int lm = lane & 15;
    const int lh = lane >> 4;

    const v8f vzero = {0.f, 0.f, 0.f, 0.f, 0.f, 0.f, 0.f, 0.f};
    v8f acc[4][2];
#pragma unroll
    for (int i = 0; i < 4; i++)
#pragma unroll
        for (int j = 0; j < 2; j++) acc[i][j] = vzero;

    // prologue: bring tile 0 into buffer 0
    issue(0, 0);
    wait_async0();
    __syncthreads();

    int buf = 0;
    for (int kt = 0; kt < Kend; kt += 64) {
        const bool more = (kt + 64) < Kend;
        if (more) issue(kt + 64, buf ^ 1);   // overlap DMA with WMMA below

#pragma unroll
        for (int ks = 0; ks < 64; ks += 32) {
            FragAB af[4], bfr[2];
#pragma unroll
            for (int mi = 0; mi < 4; mi++) {
                // A 16-bit 16x32 layout: VGPR0-3 hold K = ks + lh*8 + 0..7,
                //                        VGPR4-7 hold K = ks + 16 + lh*8 + 0..7
                const bf16* p = &Alds[buf][(wm + mi * 16 + lm) * 72 + ks + lh * 8];
                af[mi].q[0] = *(const u32x4*)p;
                af[mi].q[1] = *(const u32x4*)(p + 16);
            }
#pragma unroll
            for (int ni = 0; ni < 2; ni++) {
                // B 16-bit 32x16 layout: VGPR0-3 hold K = ks + lh*8 + 0..7,
                //                        VGPR4-7 hold K = ks + 16 + lh*8 + 0..7
                const bf16* p = &Blds[buf][(wn + ni * 16 + lm) * 72 + ks + lh * 8];
                bfr[ni].q[0] = *(const u32x4*)p;
                bfr[ni].q[1] = *(const u32x4*)(p + 16);
            }
#pragma unroll
            for (int mi = 0; mi < 4; mi++)
#pragma unroll
                for (int ni = 0; ni < 2; ni++)
                    acc[mi][ni] = __builtin_amdgcn_wmma_f32_16x16x32_bf16(
                        false, af[mi].v, false, bfr[ni].v,
                        (short)0, acc[mi][ni], false, false);
        }

        if (more) wait_async0();   // next tile landed in buf^1
        __syncthreads();
        buf ^= 1;
    }

    // --- epilogue ---
#pragma unroll
    for (int mi = 0; mi < 4; mi++) {
#pragma unroll
        for (int ni = 0; ni < 2; ni++) {
            const int ng = n0 + wn + ni * 16 + lm;
            if (ng >= N) continue;
            const float bterm = bias ? bias[ng] : 0.f;
#pragma unroll
            for (int v = 0; v < 8; v++) {
                const int mg = m0 + wm + mi * 16 + lh * 8 + v;  // C layout: M = v + lh*8
                float f = acc[mi][ni][v] * scale + bterm;
                if (act == 1) f = 0.5f * f * (1.0f + erff(f * 0.70710678118f)); // exact GELU
                const long ci = c0 + (long)mg * ldc + ng;
                if (resid) f += resid[ci];
                if (outF) outF[ci] = f;
                if (outB) outB[ci] = (bf16)f;
            }
        }
    }
}

// ---------------------------------------------------------------------------
// Helpers
// ---------------------------------------------------------------------------
__device__ inline float blk_sum(float v, float* sm) {
    int tid = threadIdx.x;
    sm[tid] = v; __syncthreads();
    for (int o = 128; o > 0; o >>= 1) { if (tid < o) sm[tid] += sm[tid + o]; __syncthreads(); }
    float r = sm[0]; __syncthreads();
    return r;
}
__device__ inline float blk_max(float v, float* sm) {
    int tid = threadIdx.x;
    sm[tid] = v; __syncthreads();
    for (int o = 128; o > 0; o >>= 1) { if (tid < o) sm[tid] = fmaxf(sm[tid], sm[tid + o]); __syncthreads(); }
    float r = sm[0]; __syncthreads();
    return r;
}

__global__ __launch_bounds__(256) void gpt_embed(
    const int* __restrict__ idx, const float* __restrict__ wte,
    const float* __restrict__ wpe, float* __restrict__ x, int E, int T, long total)
{
    long stride = (long)gridDim.x * blockDim.x;
    for (long i = (long)blockIdx.x * blockDim.x + threadIdx.x; i < total; i += stride) {
        int e = (int)(i % E);
        long bt = i / E;
        int t = (int)(bt % T);
        x[i] = wte[(long)idx[bt] * E + e] + wpe[(long)t * E + e];
    }
}

__global__ __launch_bounds__(256) void gpt_f32_to_bf16(
    const float* __restrict__ in, bf16* __restrict__ out, long n)
{
    long stride = (long)gridDim.x * blockDim.x;
    for (long i = (long)blockIdx.x * blockDim.x + threadIdx.x; i < n; i += stride)
        out[i] = (bf16)in[i];
}

// in: z * [K][N] f32  ->  out: z * [N][K] bf16
__global__ __launch_bounds__(256) void gpt_transpose_bf16(
    const float* __restrict__ in, bf16* __restrict__ out, int K, int N)
{
    long base = (long)blockIdx.z * K * N;
    long total = (long)K * N;
    long stride = (long)gridDim.x * blockDim.x;
    for (long i = (long)blockIdx.x * blockDim.x + threadIdx.x; i < total; i += stride) {
        int k = (int)(i / N);
        int n = (int)(i % N);
        out[base + (long)n * K + k] = (bf16)in[base + i];
    }
}

__global__ __launch_bounds__(256) void gpt_ln_bf16(
    const float* __restrict__ x, const float* __restrict__ g,
    const float* __restrict__ b, bf16* __restrict__ out, int E)
{
    __shared__ float sm[256];
    const long row = blockIdx.x;
    const float* xr = x + row * E;
    float s = 0.f;
    for (int i = threadIdx.x; i < E; i += 256) s += xr[i];
    s = blk_sum(s, sm);
    float mu = s / (float)E;
    float var = 0.f;
    for (int i = threadIdx.x; i < E; i += 256) { float d = xr[i] - mu; var += d * d; }
    var = blk_sum(var, sm);
    float rstd = rsqrtf(var / (float)E + 1e-5f);
    bf16* orow = out + row * E;
    for (int i = threadIdx.x; i < E; i += 256)
        orow[i] = (bf16)((xr[i] - mu) * rstd * g[i] + b[i]);
}

// causal softmax: one block per (row t, z=b*H+h); zeros above diagonal out to
// the 128-row block bound (all the clamped P@V ever reads).
__global__ __launch_bounds__(256) void gpt_softmax_causal(
    const float* __restrict__ sc, bf16* __restrict__ P, int T)
{
    __shared__ float sm[256];
    const int t = blockIdx.x;
    const long zoff = (long)blockIdx.y * T * T;
    const float* row = sc + zoff + (long)t * T;
    bf16* prow = P + zoff + (long)t * T;
    const int n = t + 1;
    const int wlen = ((t >> 7) + 1) << 7;   // row-block bound (multiple of 128)
    float mx = -1e30f;
    for (int i = threadIdx.x; i < n; i += 256) mx = fmaxf(mx, row[i]);
    mx = blk_max(mx, sm);
    float s = 0.f;
    for (int i = threadIdx.x; i < n; i += 256) s += expf(row[i] - mx);
    s = blk_sum(s, sm);
    float inv = 1.f / s;
    for (int i = threadIdx.x; i < wlen; i += 256)
        prow[i] = (bf16)((i < n) ? expf(row[i] - mx) * inv : 0.f);
}

// vT[b][h][d][s] = qkv[(b*T+s)*3E + 2E + h*D + d]   (bf16 -> bf16)
__global__ __launch_bounds__(256) void gpt_vtrans(
    const bf16* __restrict__ qkv, bf16* __restrict__ vT,
    int T, int E, int H, int D, long total)
{
    long stride = (long)gridDim.x * blockDim.x;
    for (long i = (long)blockIdx.x * blockDim.x + threadIdx.x; i < total; i += stride) {
        int s = (int)(i % T);
        long r = i / T;
        int d = (int)(r % D); r /= D;
        int h = (int)(r % H);
        int b = (int)(r / H);
        vT[i] = qkv[((long)(b * T + s)) * (3 * E) + 2 * E + h * D + d];
    }
}

// ---------------------------------------------------------------------------
extern "C" void kernel_launch(void* const* d_in, const int* in_sizes, int n_in,
                              void* d_out, int out_size, void* d_ws, size_t ws_size,
                              hipStream_t stream)
{
    (void)in_sizes; (void)n_in; (void)out_size; (void)ws_size;
    const int Bz = 2, T = 1024, E = 768, H = 12, D = 64, L = 12, V = 50257;
    const int M = Bz * T;           // 2048 rows of the residual stream
    const int Z = Bz * H;           // 24 (b,h) pairs

    const int*   idx    = (const int*)  d_in[0];
    const float* wte    = (const float*)d_in[1];
    const float* wpe    = (const float*)d_in[2];
    const float* ln1_g  = (const float*)d_in[3];
    const float* ln1_b  = (const float*)d_in[4];
    const float* attn_w = (const float*)d_in[5];
    const float* attn_b = (const float*)d_in[6];
    const float* proj_w = (const float*)d_in[7];
    const float* proj_b = (const float*)d_in[8];
    const float* ln2_g  = (const float*)d_in[9];
    const float* ln2_b  = (const float*)d_in[10];
    const float* fc_w   = (const float*)d_in[11];
    const float* fc_b   = (const float*)d_in[12];
    const float* fcp_w  = (const float*)d_in[13];
    const float* fcp_b  = (const float*)d_in[14];
    const float* lnf_g  = (const float*)d_in[15];
    const float* lnf_b  = (const float*)d_in[16];
    float* logits = (float*)d_out;

    // ---- carve workspace ----
    char* w = (char*)d_ws;
    auto carve = [&](size_t bytes) -> void* {
        void* p = (void*)w;
        w += (bytes + 255) & ~(size_t)255;
        return p;
    };
    bf16* awT   = (bf16*)carve((size_t)L * 3 * E * E * sizeof(bf16)); // [L][3E][E]
    bf16* pwT   = (bf16*)carve((size_t)L * E * E * sizeof(bf16));     // [L][E][E]
    bf16* fcwT  = (bf16*)carve((size_t)L * 4 * E * E * sizeof(bf16)); // [L][4E][E]
    bf16* fcpwT = (bf16*)carve((size_t)L * E * 4 * E * sizeof(bf16)); // [L][E][4E]
    bf16* wteb  = (bf16*)carve((size_t)V * E * sizeof(bf16));         // [V][E] (already NT)
    float* x    = (float*)carve((size_t)M * E * sizeof(float));
    bf16* xb    = (bf16*)carve((size_t)M * E * sizeof(bf16));
    bf16* qkv   = (bf16*)carve((size_t)M * 3 * E * sizeof(bf16));
    float* scor = (float*)carve((size_t)Z * T * T * sizeof(float));
    bf16* P     = (bf16*)carve((size_t)Z * T * T * sizeof(bf16));
    bf16* vT    = (bf16*)carve((size_t)Z * D * T * sizeof(bf16));
    bf16* y     = (bf16*)carve((size_t)M * E * sizeof(bf16));
    bf16* h2    = (bf16*)carve((size_t)M * 4 * E * sizeof(bf16));

    const int thr = 256;

    // ---- weight conversion (fp32 -> bf16, transposed to [N][K]) ----
    gpt_f32_to_bf16<<<4096, thr, 0, stream>>>(wte, wteb, (long)V * E);
    gpt_transpose_bf16<<<dim3(512, 1, L), thr, 0, stream>>>(attn_w, awT,  E,     3 * E);
    gpt_transpose_bf16<<<dim3(512, 1, L), thr, 0, stream>>>(proj_w, pwT,  E,     E);
    gpt_transpose_bf16<<<dim3(512, 1, L), thr, 0, stream>>>(fc_w,   fcwT, E,     4 * E);
    gpt_transpose_bf16<<<dim3(512, 1, L), thr, 0, stream>>>(fcp_w,  fcpwT, 4 * E, E);

    // ---- embedding ----
    gpt_embed<<<6144, thr, 0, stream>>>(idx, wte, wpe, x, E, T, (long)M * E);

    const long sZ = (long)T * 3 * E;   // per-batch stride inside qkv

    for (int l = 0; l < L; l++) {
        // LN1 -> xb (bf16)
        gpt_ln_bf16<<<M, thr, 0, stream>>>(x, ln1_g + l * E, ln1_b + l * E, xb, E);

        // qkv = xb @ attn_w + attn_b   -> bf16 [M, 3E]
        gpt_gemm_bf16_nt<<<dim3(18, 16, 1), thr, 0, stream>>>(
            xb, E, 0, 0,
            awT + (long)l * 3 * E * E, E, 0, 0,
            nullptr, qkv, 3 * E, 0, 0,
            nullptr, attn_b + l * 3 * E,
            M, 3 * E, E, 1.f, 0, H, 0);

        // vT[b][h][d][s]
        gpt_vtrans<<<6144, thr, 0, stream>>>(qkv, vT, T, E, H, D, (long)Z * D * T);

        // scores[z] = Q Kt / sqrt(D)   (A = Q rows, B = K rows -> NT)
        gpt_gemm_bf16_nt<<<dim3(8, 8, Z), thr, 0, stream>>>(
            qkv,       3 * E, sZ, (long)D,     // Q: +h*64
            qkv + E,   3 * E, sZ, (long)D,     // K: +E +h*64
            scor, nullptr, T, (long)H * T * T, (long)T * T,
            nullptr, nullptr,
            T, T, D, 0.125f, 0, H, 1);

        // causal softmax -> P (bf16)
        gpt_softmax_causal<<<dim3(T, Z), thr, 0, stream>>>(scor, P, T);

        // y[b,t,h*D+d] = P @ V   (B = vT rows [D][T] -> NT), causal K-limit
        gpt_gemm_bf16_nt<<<dim3(1, 8, Z), thr, 0, stream>>>(
            P,  T, (long)H * T * T, (long)T * T,
            vT, T, (long)H * D * T, (long)D * T,
            nullptr, y, E, (long)T * E, (long)D,
            nullptr, nullptr,
            T, D, T, 1.f, 0, H, 2);

        // x = x + y @ proj_w + proj_b
        gpt_gemm_bf16_nt<<<dim3(6, 16, 1), thr, 0, stream>>>(
            y, E, 0, 0,
            pwT + (long)l * E * E, E, 0, 0,
            x, nullptr, E, 0, 0,
            x, proj_b + l * E,
            M, E, E, 1.f, 0, H, 0);

        // LN2 -> xb
        gpt_ln_bf16<<<M, thr, 0, stream>>>(x, ln2_g + l * E, ln2_b + l * E, xb, E);

        // h2 = gelu(xb @ fc_w + fc_b)  -> bf16 [M, 4E]
        gpt_gemm_bf16_nt<<<dim3(24, 16, 1), thr, 0, stream>>>(
            xb, E, 0, 0,
            fcwT + (long)l * 4 * E * E, E, 0, 0,
            nullptr, h2, 4 * E, 0, 0,
            nullptr, fc_b + l * 4 * E,
            M, 4 * E, E, 1.f, 1, H, 0);

        // x = x + h2 @ fcp_w + fcp_b
        gpt_gemm_bf16_nt<<<dim3(6, 16, 1), thr, 0, stream>>>(
            h2, 4 * E, 0, 0,
            fcpwT + (long)l * E * 4 * E, 4 * E, 0, 0,
            x, nullptr, E, 0, 0,
            x, fcp_b + l * E,
            M, E, 4 * E, 1.f, 0, H, 0);
    }

    // final LN -> xb
    gpt_ln_bf16<<<M, thr, 0, stream>>>(x, lnf_g, lnf_b, xb, E);

    // logits = xb @ wte^T   (wte is [V][E] -> already NT)
    gpt_gemm_bf16_nt<<<dim3((V + 127) / 128, 16, 1), thr, 0, stream>>>(
        xb, E, 0, 0,
        wteb, E, 0, 0,
        logits, nullptr, V, 0, 0,
        nullptr, nullptr,
        M, V, E, 1.f, 0, H, 0);
}
